// CrystalEncoder_11570641896052
// MI455X (gfx1250) — matrix operand order repaired and checked by reading, not compiled, not run on hardware
//
#include <hip/hip_runtime.h>

// ---------------------------------------------------------------------------
// CrystalEncoder on MI455X (gfx1250): gate GEMM on v_wmma_f32_16x16x32_f16.
// Shapes: B=8, N=256, H=128, LAT=64, NL=2, BINS=40 (K padded to 64).
// ---------------------------------------------------------------------------

typedef _Float16 v16h __attribute__((ext_vector_type(16)));
typedef float    v8f  __attribute__((ext_vector_type(8)));

#define Bb    8
#define Nn    256
#define Hh    128
#define LATd  64
#define NLn   2
#define BINSn 40
#define KPAD  64
#define HSTR  136   // f16 LDS stride for staged h: (j+8)*136 vs j*136 differ by
                    // 32 banks after >>1, so D-layout readback is conflict-free

#define GAMMA 25.0f              // 1/(VMAX/BINS)^2 = 1/0.2^2
#define CSTEP (8.0f / 39.0f)     // linspace(0, 8, 40) spacing
#define LOG2E 1.44269504088896f
#define LN2   0.69314718055995f

// raw-instruction transcendentals (avoid host-shadowed __exp2f/__log2f)
__device__ __forceinline__ float exp2_raw(float x) { return __builtin_amdgcn_exp2f(x); } // v_exp_f32
__device__ __forceinline__ float log2_raw(float x) { return __builtin_amdgcn_logf(x); }  // v_log_f32

// branchless, overflow-safe softplus: max(x,0) + ln2*log2(1+2^(-|x|*log2e))
__device__ __forceinline__ float softplus_fast(float x) {
  float t = exp2_raw(-LOG2E * __builtin_fabsf(x));    // t in (0,1]
  return fmaxf(x, 0.0f) + LN2 * log2_raw(1.0f + t);
}

// --- prep: cart = frac @ lattice (per-batch 3x3) ---------------------------
__global__ void crystal_prep(const float* __restrict__ frac,
                             const float* __restrict__ lat,
                             float* __restrict__ cart) {
  int idx = blockIdx.x * blockDim.x + threadIdx.x;   // b*N + n
  if (idx >= Bb * Nn) return;
  int b = idx / Nn;
  const float* f = frac + idx * 3;
  const float* L = lat + b * 9;
#pragma unroll
  for (int e = 0; e < 3; ++e)
    cart[idx * 3 + e] = f[0] * L[e] + f[1] * L[3 + e] + f[2] * L[6 + e];
}

// --- prep: h0 = emb_table[atom_types]  (row 0 of table is zeros) -----------
__global__ void crystal_embed(const int* __restrict__ at,
                              const float* __restrict__ emb,
                              float* __restrict__ h0) {
  int idx = blockIdx.x * blockDim.x + threadIdx.x;   // over B*N*H
  if (idx >= Bb * Nn * Hh) return;
  int bn = idx >> 7;
  int hc = idx & (Hh - 1);
  h0[idx] = emb[at[bn] * Hh + hc];
}

// --- one message-passing layer: WG = one (b,i) row -------------------------
// blockDim = 256 (8 waves). Wave w owns h-columns [16w, 16w+16).
__global__ void crystal_layer(const int*   __restrict__ at,
                              const float* __restrict__ cart,
                              const float* __restrict__ h_in,
                              float*       __restrict__ h_out,
                              const float* __restrict__ edge_w,   // (BINS,H)
                              const float* __restrict__ edge_b,   // (H)
                              const float* __restrict__ node_w,   // (H,H)
                              const float* __restrict__ node_b) { // (H)
  extern __shared__ char smem[];
  _Float16* rbfs = (_Float16*)smem;                 // [Nn][KPAD]  32 KB
  _Float16* hsh  = rbfs + Nn * KPAD;                // [Nn][HSTR]  69.6 KB
  float*    agg  = (float*)(hsh + Nn * HSTR);       // [Hh]        512 B

  const int tid = threadIdx.x;
  const int b   = blockIdx.x / Nn;
  const int i   = blockIdx.x % Nn;

  // ---- stage h[b,:,:] to LDS as f16 (padded stride) ----
  for (int idx = tid; idx < Nn * Hh; idx += 256) {
    int n  = idx >> 7;
    int hc = idx & (Hh - 1);
    hsh[n * HSTR + hc] = (_Float16)h_in[(b * Nn + n) * Hh + hc];
  }

  // ---- rbf row for atom i: thread t computes all bins for j = t ----
  {
    const int j = tid;
    const float* ci = cart + (b * Nn + i) * 3;
    const float* cj = cart + (b * Nn + j) * 3;
    float dx = ci[0] - cj[0], dy = ci[1] - cj[1], dz = ci[2] - cj[2];
    float d  = sqrtf(dx * dx + dy * dy + dz * dz + 1e-6f);
#pragma unroll 8
    for (int k = 0; k < KPAD; ++k) {
      float v = 0.0f;
      if (k < BINSn) {
        float t = d - (float)k * CSTEP;
        v = exp2_raw(-GAMMA * LOG2E * t * t);
      }
      rbfs[j * KPAD + k] = (_Float16)v;
    }
  }
  __syncthreads();

  // ---- per-wave WMMA gate + on-the-fly softplus * h_j reduction ----
  const int lane  = tid & 31;
  const int wave  = tid >> 5;
  const int hcol  = wave * 16 + (lane & 15);       // N-index of this lane
  const int half  = (lane < 16) ? 0 : 8;           // upper half-wave offset
  const int kbase = (lane < 16) ? 0 : 16;          // unused; tiles take K base 0/8 from half

  // B tiles (invariant over j): element e of v16h = K (half+e+8*(e>=8)) at column hcol.
  v16h bt0, bt1;
#pragma unroll
  for (int e = 0; e < 16; ++e) {
    int k0 = half + e + ((e >> 3) << 3);
    int k1 = 32 + half + e + ((e >> 3) << 3);
    bt0[e] = (k0 < BINSn) ? (_Float16)edge_w[k0 * Hh + hcol] : (_Float16)0.0f;
    bt1[e] = (k1 < BINSn) ? (_Float16)edge_w[k1 * Hh + hcol] : (_Float16)0.0f;
  }
  const float eb = edge_b[hcol];

  // hoisted per-lane LDS bases (constant offsets inside the loop)
  const _Float16* arow  = &rbfs[(lane & 15) * KPAD + half];   // A-tile row base
  const _Float16* hbase = &hsh[half * HSTR + hcol];           // h_j read base

  float acc = 0.0f;
#pragma unroll 4
  for (int jt = 0; jt < 16; ++jt) {
    // A tile: lane<16 -> row j = 16jt+lane, K {0..7,16..23}(+kstep);
    //         lane>=16 -> row j = 16jt+lane-16, K {8..15,24..31}(+kstep).
    const _Float16* ar = arow + jt * (16 * KPAD);
    v16h a0, a1;
#pragma unroll
    for (int e = 0; e < 8; ++e) {
      a0[e]     = ar[e];
      a0[e + 8] = ar[16 + e];
      a1[e]     = ar[32 + e];
      a1[e + 8] = ar[48 + e];
    }
    // preload the 8 h_j values (fills the WMMA->VALU hazard window)
    const _Float16* hp = hbase + jt * (16 * HSTR);
    float hv[8];
#pragma unroll
    for (int v = 0; v < 8; ++v) hv[v] = (float)hp[v * HSTR];

    v8f c = {};
    c = __builtin_amdgcn_wmma_f32_16x16x32_f16(false, a0, false, bt0,
                                               (short)0, c, false, false);
    c = __builtin_amdgcn_wmma_f32_16x16x32_f16(false, a1, false, bt1,
                                               (short)0, c, false, false);
    // D layout: VGPR v holds row M = v (+8 for upper half-wave), col = hcol.
#pragma unroll
    for (int v = 0; v < 8; ++v)
      acc = fmaf(softplus_fast(c[v] + eb), hv[v], acc);  // h_j == 0 if padded
  }
  // fold the two D-row halves (lane L and L+16 share the same h column)
  acc += __shfl_xor(acc, 16, 32);
  if (lane < 16) agg[hcol] = acc;
  __syncthreads();

  // ---- node update: h = (h + silu(agg @ node_w + node_b)) * mask_i ----
  if (tid < Hh) {
    const int hc = tid;
    float dot = node_b[hc];
#pragma unroll 8
    for (int k = 0; k < Hh; ++k) dot = fmaf(agg[k], node_w[k * Hh + hc], dot);
    // silu: x * sigmoid(x), with raw v_exp/v_rcp
    float s = dot * __builtin_amdgcn_rcpf(1.0f + exp2_raw(-LOG2E * dot));
    float hold = h_in[(b * Nn + i) * Hh + hc];
    float outv = (at[b * Nn + i] != 0) ? (hold + s) : 0.0f;
    h_out[(b * Nn + i) * Hh + hc] = outv;
  }
}

// --- masked mean pool + mu / log_var heads ---------------------------------
__global__ void crystal_pool(const int*   __restrict__ at,
                             const float* __restrict__ h,
                             const float* __restrict__ mu_w,
                             const float* __restrict__ mu_b,
                             const float* __restrict__ var_w,
                             const float* __restrict__ var_b,
                             float* __restrict__ out) {
  __shared__ float part[2][Hh];
  __shared__ float g[Hh];
  __shared__ int   cnt;
  const int b   = blockIdx.x;
  const int tid = threadIdx.x;
  if (tid == 0) cnt = 0;
  __syncthreads();
  if (tid < Nn && at[b * Nn + tid] != 0) atomicAdd(&cnt, 1);

  // invalid rows are exactly zero, so plain sum == masked sum
  const int sl = tid >> 7, hc = tid & (Hh - 1);
  float s = 0.0f;
  for (int n = sl * 128; n < sl * 128 + 128; ++n)
    s += h[(b * Nn + n) * Hh + hc];
  part[sl][hc] = s;
  __syncthreads();
  if (tid < Hh) g[tid] = (part[0][tid] + part[1][tid]) / ((float)cnt + 1e-6f);
  __syncthreads();

  if (tid < LATd) {
    float d = mu_b[tid];
#pragma unroll 8
    for (int k = 0; k < Hh; ++k) d = fmaf(g[k], mu_w[k * LATd + tid], d);
    out[b * LATd + tid] = d;
  } else if (tid < 2 * LATd) {
    int l = tid - LATd;
    float d = var_b[l];
#pragma unroll 8
    for (int k = 0; k < Hh; ++k) d = fmaf(g[k], var_w[k * LATd + l], d);
    out[Bb * LATd + b * LATd + l] = d;
  }
}

// ---------------------------------------------------------------------------
extern "C" void kernel_launch(void* const* d_in, const int* in_sizes, int n_in,
                              void* d_out, int out_size, void* d_ws, size_t ws_size,
                              hipStream_t stream) {
  const int*   at     = (const int*)d_in[0];    // (B,N) int32; 0 == padding
  const float* frac   = (const float*)d_in[1];  // (B,N,3)
  const float* lat    = (const float*)d_in[2];  // (B,3,3)
  /* d_in[3] = mask (bool) — derived from atom_types instead */
  const float* emb    = (const float*)d_in[4];  // (100,H)
  const float* edge_w = (const float*)d_in[5];  // (NL,BINS,H)
  const float* edge_b = (const float*)d_in[6];  // (NL,H)
  const float* node_w = (const float*)d_in[7];  // (NL,H,H)
  const float* node_b = (const float*)d_in[8];  // (NL,H)
  const float* mu_w   = (const float*)d_in[9];  // (H,LAT)
  const float* mu_b   = (const float*)d_in[10]; // (LAT)
  const float* var_w  = (const float*)d_in[11]; // (H,LAT)
  const float* var_b  = (const float*)d_in[12]; // (LAT)
  float* out = (float*)d_out;

  // workspace: cart (B,N,3) | hA (B,N,H) | hB (B,N,H)  (~2.1 MB total)
  float* cart = (float*)d_ws;
  float* hA   = cart + Bb * Nn * 3;
  float* hB   = hA + Bb * Nn * Hh;

  crystal_prep<<<(Bb * Nn + 255) / 256, 256, 0, stream>>>(frac, lat, cart);
  crystal_embed<<<(Bb * Nn * Hh + 255) / 256, 256, 0, stream>>>(at, emb, hA);

  const size_t lds = (size_t)(Nn * KPAD + Nn * HSTR) * sizeof(_Float16)
                   + (size_t)Hh * sizeof(float);     // ~102.9 KB

  // layer 0: hA -> hB
  crystal_layer<<<Bb * Nn, 256, lds, stream>>>(
      at, cart, hA, hB, edge_w, edge_b, node_w, node_b);
  // layer 1: hB -> hA
  crystal_layer<<<Bb * Nn, 256, lds, stream>>>(
      at, cart, hB, hA,
      edge_w + BINSn * Hh, edge_b + Hh, node_w + Hh * Hh, node_b + Hh);

  crystal_pool<<<Bb, 256, 0, stream>>>(at, hA, mu_w, mu_b, var_w, var_b, out);
}
